// Model_51599737094550
// MI455X (gfx1250) — matrix operand order repaired and checked by reading, not compiled, not run on hardware
//
#include <hip/hip_runtime.h>

typedef __attribute__((ext_vector_type(16))) _Float16 v16h;
typedef __attribute__((ext_vector_type(8)))  float    v8f;

#define NDIM 100
#define PDIM 12
#define BDIM 128
#define NN   10000   // N*N
#define NPAD 112     // 7 tiles of 16 rows
#define MPAD 128     // 4 K-chunks of 32

#if __has_builtin(__builtin_amdgcn_exp2f)
#define EXP2(x) __builtin_amdgcn_exp2f(x)   // native v_exp_f32 (exp2)
#else
#define EXP2(x) exp2f(x)
#endif
#define LOG2E 1.44269504088896340736f

// ---------------- Phase A: F = g @ (w*w)  (bandwidth bound: 400MB @ 23.3TB/s) ----
__global__ __launch_bounds__(256) void matvec_F_kernel(const float* __restrict__ g,
                                                       const float* __restrict__ w,
                                                       float* __restrict__ F_ws,
                                                       float* __restrict__ F_out) {
  const int row = blockIdx.x;
  const float4* gr = (const float4*)(g + (size_t)row * NN);
  const float4* w4 = (const float4*)w;
  float acc = 0.f;
  for (int j = threadIdx.x; j < NN / 4; j += 256) {
    __builtin_prefetch(&gr[j + 512], 0, 0);   // global_prefetch_b8, speculative
    float4 gv = gr[j];
    float4 wv = w4[j];
    acc += gv.x * wv.x * wv.x + gv.y * wv.y * wv.y +
           gv.z * wv.z * wv.z + gv.w * wv.w * wv.w;
  }
  __shared__ float red[256];
  red[threadIdx.x] = acc;
  __syncthreads();
  for (int s = 128; s > 0; s >>= 1) {
    if (threadIdx.x < s) red[threadIdx.x] += red[threadIdx.x + s];
    __syncthreads();
  }
  if (threadIdx.x == 0) { F_ws[row] = red[0]; F_out[row] = red[0]; }
}

// ---------------- Phase A2: per-row min of F (softmax stabilizer for ALL alphas) --
__global__ void fmin_kernel(const float* __restrict__ F_ws, float* __restrict__ Fmin) {
  int n = threadIdx.x;
  if (n < NDIM) {
    float m = F_ws[n * NDIM];
    for (int j = 1; j < NDIM; ++j) m = fminf(m, F_ws[n * NDIM + j]);
    Fmin[n] = m;
  }
}

// ---------------- Phase B: gathered on-the-fly softmax matvec via WMMA ------------
// One block per b; wave w owns n-rows [16w,16w+16). For each p (fixed order, det.):
//   A[16x32] : row0 = x[b,m,p] chunk, row1 = 1.0, rows 2..15 = 0
//   B[32x16] : B[m,ncol] = exp2(s2*(F[n,m]-Fmin[n])), s2 = -alpha*log2e
//   D += A@B over 4 K-chunks  ->  D[0,n]=numerator, D[1,n]=denominator
// Lane L<16 owns n=n0+L: zacc += D0/D1. Branch-free inner loop, float4 LDS reads.
__global__ __launch_bounds__(256) void softmax_z_kernel(const float* __restrict__ x,
                                                        const float* __restrict__ alphas,
                                                        const int*   __restrict__ x_i,
                                                        const float* __restrict__ F_ws,
                                                        const float* __restrict__ Fmin,
                                                        float* __restrict__ Z) {
  __shared__ float fs2[NPAD * MPAD];   // (F[n,m]-Fmin[n]), 1e30 padding -> exp2()==0
  __shared__ float xsT[PDIM * MPAD];   // x transposed per-p rows, zero padded
  const int b   = blockIdx.x;
  const int tid = threadIdx.x;
  for (int i = tid; i < NPAD * MPAD; i += 256) {
    int n = i >> 7, m = i & (MPAD - 1);
    fs2[i] = (n < NDIM && m < NDIM) ? (F_ws[n * NDIM + m] - Fmin[n]) : 1e30f;
  }
  for (int i = tid; i < PDIM * MPAD; i += 256) {
    int p = i >> 7, m = i & (MPAD - 1);
    xsT[i] = (m < NDIM) ? x[(b * NDIM + m) * PDIM + p] : 0.f;
  }
  __syncthreads();

  const int wave = tid >> 5;
  const int lane = tid & 31;
  if (wave >= 7) return;               // 7 tiles cover n = 0..111 (>=100 padded)

  const int n0     = wave * 16;
  const int laneHi = lane >> 4;
  const int laneLo = lane & 15;
  const int nB     = n0 + laneLo;      // B-matrix column (n) this lane produces
  const float selX   = (laneLo == 0) ? 1.f : 0.f;   // A row 0: x values
  const float selOne = (laneLo == 1) ? 1.f : 0.f;   // A row 1: ones
  const bool  valid  = (laneHi == 0) && (nB < NDIM);

  const float4* fs2v = (const float4*)fs2;
  const float4* xsv  = (const float4*)xsT;

  float zacc = 0.f;

#pragma unroll 2
  for (int p = 0; p < PDIM; ++p) {
    // s2 uniform across the block: one scalar load + SALU per p
    const float s2 = -alphas[x_i[b * PDIM + p]] * LOG2E;
    v8f c = {};
#pragma unroll
    for (int kk = 0; kk < 4; ++kk) {
      // ---- B tile (exp side): element e <-> K = e + 16*laneHi, m = 32*kk + K.
      // 16 consecutive m values => 4x float4 LDS loads; mul + native exp2 + cvt.
      v16h bm;
      const int eb = nB * (MPAD / 4) + 8 * kk + 4 * laneHi;
#pragma unroll
      for (int q = 0; q < 4; ++q) {
        float4 f4 = fs2v[eb + q];
        bm[4 * q + 0] = (_Float16)EXP2(s2 * f4.x);
        bm[4 * q + 1] = (_Float16)EXP2(s2 * f4.y);
        bm[4 * q + 2] = (_Float16)EXP2(s2 * f4.z);
        bm[4 * q + 3] = (_Float16)EXP2(s2 * f4.w);
      }
      // ---- A tile (x/ones side): e <-> m = 32kk + 16*laneHi + e.
      v16h a;
      const int xa = p * (MPAD / 4) + 8 * kk + 4 * laneHi;
#pragma unroll
      for (int q = 0; q < 2; ++q) {
        float4 x0 = xsv[xa + q];           // run 0: e = 4q .. 4q+3
        a[4 * q + 0] = (_Float16)(x0.x * selX + selOne);
        a[4 * q + 1] = (_Float16)(x0.y * selX + selOne);
        a[4 * q + 2] = (_Float16)(x0.z * selX + selOne);
        a[4 * q + 3] = (_Float16)(x0.w * selX + selOne);
        float4 x1 = xsv[xa + 2 + q];       // run 1: e = 8+4q .. 8+4q+3
        a[8 + 4 * q + 0] = (_Float16)(x1.x * selX + selOne);
        a[8 + 4 * q + 1] = (_Float16)(x1.y * selX + selOne);
        a[8 + 4 * q + 2] = (_Float16)(x1.z * selX + selOne);
        a[8 + 4 * q + 3] = (_Float16)(x1.w * selX + selOne);
      }
      c = __builtin_amdgcn_wmma_f32_16x16x32_f16(false, a, false, bm,
                                                 (short)0, c, false, false);
    }
    // D[0,n] = numerator, D[1,n] = denominator, both in lanes 0..15 (den >= 1).
    const float upd = c[0] / c[1];
    zacc += valid ? upd : 0.f;
  }
  if (valid) Z[b * NDIM + nB] = zacc;
}

extern "C" void kernel_launch(void* const* d_in, const int* in_sizes, int n_in,
                              void* d_out, int out_size, void* d_ws, size_t ws_size,
                              hipStream_t stream) {
  const float* x      = (const float*)d_in[0];   // [128,100,12]
  const float* g      = (const float*)d_in[1];   // [10000,10000]
  const float* w      = (const float*)d_in[2];   // [10000,1]
  const float* alphas = (const float*)d_in[3];   // [4000,1]
  const int*   x_i    = (const int*)  d_in[4];   // [128,12]

  float* Z     = (float*)d_out;                  // [128,100]
  float* F_out = (float*)d_out + BDIM * NDIM;    // [10000]
  float* F_ws  = (float*)d_ws;                   // 10000 floats
  float* FminW = F_ws + NN;                      // 100 floats

  matvec_F_kernel<<<NN, 256, 0, stream>>>(g, w, F_ws, F_out);
  fmin_kernel<<<1, 128, 0, stream>>>(F_ws, FminW);
  softmax_z_kernel<<<BDIM, 256, 0, stream>>>(x, alphas, x_i, F_ws, FminW, Z);
}
